// Decoder_26585847562664
// MI455X (gfx1250) — matrix operand order repaired and checked by reading, not compiled, not run on hardware
//
#include <hip/hip_runtime.h>

// B=2, S=2048, D=1024, H=16, HD=64.
// Pipeline:
//   1) qkv_proj_kernel: Q/K/V = X @ W^T + b via v_wmma_f32_16x16x32_f16.
//      Global->LDS staging uses the Tensor Data Mover (tensor_load_to_lds, TENSORcnt),
//      double-buffered, with TDM-injected LDS padding (34-dword row stride, bank-conflict free).
//      Results stored f16 in workspace: Q,K as [b,h,s,hd]; V transposed [b,h,hd,s].
//   2) attn_kernel: flash-attention (online softmax), one wave per 32 query rows,
//      QK^T and PV on WMMA, global_prefetch for the next key tile, fp32 output.
// Workspace use: 3 * B*H*S*HD f16 = 24 MB.

typedef __attribute__((ext_vector_type(16))) _Float16 v16h;
typedef __attribute__((ext_vector_type(8)))  _Float16 v8h;
typedef __attribute__((ext_vector_type(2)))  _Float16 v2h;
typedef __attribute__((ext_vector_type(8)))  float    v8f;
typedef __attribute__((ext_vector_type(2)))  float    v2f;
typedef __attribute__((ext_vector_type(4)))  unsigned int u32x4;
typedef __attribute__((ext_vector_type(8)))  int      i32x8;
typedef __attribute__((ext_vector_type(4)))  int      i32x4;

static constexpr int Bb = 2, Ss = 2048, Dd = 1024, Hh = 16, HDd = 64;

// LDS layout for the projection kernel (bytes), per double-buffer:
//   A tile: 128 rows x 34 dwords (32 data + 2 TDM pad)  = 17408 B
//   B tile:  64 rows x 34 dwords                         =  8704 B
static constexpr unsigned A_TILE_BYTES = 128 * 34 * 4;
static constexpr unsigned B_TILE_BYTES = 64 * 34 * 4;
static constexpr unsigned BUF_BYTES    = A_TILE_BYTES + B_TILE_BYTES;  // 26112
static constexpr unsigned PROJ_LDS     = 2 * BUF_BYTES;                // 52224

#define WMMA_F16(a, b, c) \
    __builtin_amdgcn_wmma_f32_16x16x32_f16(false, (a), false, (b), (short)0, (c), false, false)

__device__ inline v8f vzero8() {
    v8f z;
#pragma unroll
    for (int i = 0; i < 8; ++i) z[i] = 0.0f;
    return z;
}

// ---------------------------------------------------------------------------
// TDM: issue a 2-D tensor tile load (4-byte elements) into LDS.
// D# packing per CDNA5 ISA (groups 0/1; groups 2/3 zero for 2-D).
// tile rows are 32 dwords; pad_enable with pad_interval=4 (32 dwords) and
// pad_amount=1 (2 dwords) -> LDS row stride 34 dwords.
// Must be executed by a uniform (whole-wave) branch: EXEC is ignored, op is per-wave.
// This toolchain exposes the 6-arg builtin:
//   (uint32x4 g0, int32x8 g1, int32x4 g2, int32x4 g3, int32x8 g4, i32 cpol)
// ---------------------------------------------------------------------------
__device__ inline void tdm_load_tile_f32(unsigned lds_byte_off, const float* gptr,
                                         unsigned tensor_d0, unsigned tensor_d1,
                                         unsigned stride0,
                                         unsigned tile_d0, unsigned tile_d1) {
    const unsigned long long ga = (unsigned long long)gptr;
    u32x4 g0;
    g0[0] = 1u;                                  // count=1 (valid), user mode, no gather
    g0[1] = lds_byte_off;                        // lds_addr [63:32]
    g0[2] = (unsigned)(ga & 0xffffffffu);        // global_addr[31:0]  -> bits 95:64
    g0[3] = (unsigned)((ga >> 32) & 0x01ffffffu) // global_addr[56:32] -> bits 120:96
          | (2u << 30);                          // type=2 ("image")   -> bits 127:126

    i32x8 g1;
    g1[0] = (int)((2u << 16)                     // data_size=2 -> 4 bytes
                | (1u << 20)                     // pad_enable
                | (4u << 22)                     // pad_interval: 32 dwords
                | (1u << 25));                   // pad_amount: 2 dwords
    g1[1] = (int)((tensor_d0 & 0xffffu) << 16);  // atomic_barrier_addr=0 | dim0 lo16
    g1[2] = (int)(((tensor_d0 >> 16) & 0xffffu) | ((tensor_d1 & 0xffffu) << 16));
    g1[3] = (int)(((tensor_d1 >> 16) & 0xffffu) | ((tile_d0 & 0xffffu) << 16));
    g1[4] = (int)(tile_d1 & 0xffffu);            // tile_dim1 | tile_dim2=0
    g1[5] = (int)stride0;                        // tensor_dim0_stride lo32
    g1[6] = 0;                                   // stride hi16 | dim1_stride lo16
    g1[7] = 0;

    i32x4 z4;
    z4[0] = 0; z4[1] = 0; z4[2] = 0; z4[3] = 0;
    i32x8 z8;
#pragma unroll
    for (int i = 0; i < 8; ++i) z8[i] = 0;
    __builtin_amdgcn_tensor_load_to_lds(g0, g1, z4, z4, z8, 0);
}

// A-fragment (16x32 f16) from row-major f32 LDS tile, row stride ld (dwords).
// Contiguous k order (same as B): lane m = L&15, g = L>>4; VGPR j holds K = g*16 + 2*j (+1).
__device__ inline v16h load_a_frag_f32(const float* base, int ld) {
    const int lane = threadIdx.x & 31;
    const int m = lane & 15, g = lane >> 4;
    v16h a;
#pragma unroll
    for (int j = 0; j < 8; ++j) {
        const int k = (g << 4) + (j << 1);
        v2f p = *(const v2f*)(base + m * ld + k);
        a[2 * j]     = (_Float16)p[0];
        a[2 * j + 1] = (_Float16)p[1];
    }
    return a;
}

// B-fragment (32x16 f16) from N-major f32 LDS tile src[n][k], row stride ld (dwords).
// Contiguous k order (same as A): lane n = L&15, g = L>>4; element e holds K = g*16 + e.
__device__ inline v16h load_b_frag_f32(const float* base, int ld) {
    const int lane = threadIdx.x & 31;
    const int n = lane & 15, g = lane >> 4;
    const float* p = base + n * ld + g * 16;
    v16h b;
#pragma unroll
    for (int e = 0; e < 16; ++e) b[e] = (_Float16)p[e];
    return b;
}

// A-fragment (16x32 f16) from row-major f16 storage, row stride ld (halves).
__device__ inline v16h load_a_frag_f16(const _Float16* base, int ld) {
    const int lane = threadIdx.x & 31;
    const int m = lane & 15, g = lane >> 4;
    v16h a;
#pragma unroll
    for (int j = 0; j < 8; ++j) {
        const int k = (g << 4) + (j << 1);
        v2h p = *(const v2h*)(base + m * ld + k);
        a[2 * j] = p[0];
        a[2 * j + 1] = p[1];
    }
    return a;
}

// B-fragment (32x16 f16) from N-major f16 storage, 16B-aligned rows: two b128 loads.
__device__ inline v16h load_b_frag_vec(const _Float16* base, long ld) {
    const int lane = threadIdx.x & 31;
    const int n = lane & 15, g = lane >> 4;
    const _Float16* p = base + (long)n * ld + g * 16;
    v8h lo = *(const v8h*)(p);
    v8h hi = *(const v8h*)(p + 8);
    v16h b;
#pragma unroll
    for (int e = 0; e < 8; ++e) { b[e] = lo[e]; b[e + 8] = hi[e]; }
    return b;
}

// ---------------------------------------------------------------------------
// Kernel 1: Y = X @ W^T + bias, f16 output. grid=(M/128, N/64, 3), block=256.
// TDM double-buffered staging; wave 0 drives the DMA, all 8 waves compute.
// ---------------------------------------------------------------------------
__global__ __launch_bounds__(256)
void qkv_proj_kernel(const float* __restrict__ qx, const float* __restrict__ kx,
                     const float* __restrict__ vx,
                     const float* __restrict__ Wq, const float* __restrict__ bq,
                     const float* __restrict__ Wk, const float* __restrict__ bk,
                     const float* __restrict__ Wv, const float* __restrict__ bv,
                     _Float16* __restrict__ q16, _Float16* __restrict__ k16,
                     _Float16* __restrict__ v16t) {
    extern __shared__ char smem[];  // 2 x (A tile + B tile), dynamic => LDS offset 0

    const int z = blockIdx.z;
    const float* X    = (z == 0) ? qx : (z == 1) ? kx : vx;
    const float* W    = (z == 0) ? Wq : (z == 1) ? Wk : Wv;
    const float* bias = (z == 0) ? bq : (z == 1) ? bk : bv;

    const int t = threadIdx.x;
    const int lane = t & 31, wid = t >> 5;
    const int waveM = wid & 3, waveN = wid >> 2;    // 4x2 wave grid -> 128x64 tile
    const int m0 = blockIdx.x * 128, n0 = blockIdx.y * 64;

    v8f acc[2][2];
#pragma unroll
    for (int mi = 0; mi < 2; ++mi)
#pragma unroll
        for (int ni = 0; ni < 2; ++ni) acc[mi][ni] = vzero8();

    const int NK = Dd / 32;  // 32 K-tiles

    // Prologue: wave 0 issues the first tile's DMAs.
    if (wid == 0) {
        tdm_load_tile_f32(0,            X + (long)m0 * Dd, Dd, Bb * Ss, Dd, 32, 128);
        tdm_load_tile_f32(A_TILE_BYTES, W + (long)n0 * Dd, Dd, Dd,      Dd, 32, 64);
    }

    int buf = 0;
    for (int it = 0; it < NK; ++it) {
        if (wid == 0) __builtin_amdgcn_s_wait_tensorcnt(0);
        __syncthreads();  // current buffer visible to all waves

        // Issue next tile into the other buffer while we compute this one.
        if (wid == 0 && it + 1 < NK) {
            const int k1 = (it + 1) * 32;
            const unsigned off = (unsigned)((buf ^ 1) * BUF_BYTES);
            tdm_load_tile_f32(off,                X + (long)m0 * Dd + k1, Dd, Bb * Ss, Dd, 32, 128);
            tdm_load_tile_f32(off + A_TILE_BYTES, W + (long)n0 * Dd + k1, Dd, Dd,      Dd, 32, 64);
        }

        const float* As = (const float*)(smem + buf * BUF_BYTES);
        const float* Bs = (const float*)(smem + buf * BUF_BYTES + A_TILE_BYTES);

        v16h aF[2], bF[2];
        aF[0] = load_a_frag_f32(As + (waveM * 32 + 0)  * 34, 34);
        aF[1] = load_a_frag_f32(As + (waveM * 32 + 16) * 34, 34);
        bF[0] = load_b_frag_f32(Bs + (waveN * 32 + 0)  * 34, 34);
        bF[1] = load_b_frag_f32(Bs + (waveN * 32 + 16) * 34, 34);
#pragma unroll
        for (int mi = 0; mi < 2; ++mi)
#pragma unroll
            for (int ni = 0; ni < 2; ++ni) acc[mi][ni] = WMMA_F16(aF[mi], bF[ni], acc[mi][ni]);

        __syncthreads();  // everyone done with buf before it is refilled next round
        buf ^= 1;
    }

    // Epilogue: add bias, convert to f16, scatter to [b,h,s,hd] (or [b,h,hd,s] for V).
    const int g = lane >> 4, nl = lane & 15;
#pragma unroll
    for (int mi = 0; mi < 2; ++mi) {
#pragma unroll
        for (int ni = 0; ni < 2; ++ni) {
            const int ng = n0 + waveN * 32 + ni * 16 + nl;
            const float bval = bias[ng];
            const int hh = ng >> 6, hd = ng & 63;
#pragma unroll
            for (int r = 0; r < 8; ++r) {
                const int mg = m0 + waveM * 32 + mi * 16 + r + g * 8;
                const int bi = mg >> 11, si = mg & 2047;
                const _Float16 hv = (_Float16)(acc[mi][ni][r] + bval);
                if (z == 0)
                    q16[(((long)bi * Hh + hh) * Ss + si) * HDd + hd] = hv;
                else if (z == 1)
                    k16[(((long)bi * Hh + hh) * Ss + si) * HDd + hd] = hv;
                else
                    v16t[(((long)bi * Hh + hh) * HDd + hd) * Ss + si] = hv;
            }
        }
    }
}

// ---------------------------------------------------------------------------
// Kernel 2: flash attention. grid=(S/32, H, B), block=32 (one wave).
// ---------------------------------------------------------------------------
__global__ __launch_bounds__(32)
void attn_kernel(const _Float16* __restrict__ q16, const _Float16* __restrict__ k16,
                 const _Float16* __restrict__ v16t, float* __restrict__ out) {
    const int lane = threadIdx.x;
    const int q0 = blockIdx.x * 32;
    const int h = blockIdx.y, b = blockIdx.z;
    const long bh = (long)b * Hh + h;

    __shared__ _Float16 Ps[32 * 34];  // probability tile, C-layout -> A-layout round trip

    const int g = lane >> 4, nl = lane & 15;

    // Q A-fragments: [mi][k-chunk], loaded once.
    v16h aQ[2][2];
#pragma unroll
    for (int mi = 0; mi < 2; ++mi)
#pragma unroll
        for (int kc = 0; kc < 2; ++kc)
            aQ[mi][kc] = load_a_frag_f16(q16 + (bh * Ss + q0 + mi * 16) * HDd + kc * 32, HDd);

    v8f o[2][4];
#pragma unroll
    for (int mi = 0; mi < 2; ++mi)
#pragma unroll
        for (int nv = 0; nv < 4; ++nv) o[mi][nv] = vzero8();

    float mrun[2][8], lrun[2][8];
#pragma unroll
    for (int mi = 0; mi < 2; ++mi)
#pragma unroll
        for (int r = 0; r < 8; ++r) { mrun[mi][r] = -3.0e38f; lrun[mi][r] = 0.0f; }

    const float scale = 0.03125f;  // 1/sqrt(d_model) = 1/32

    for (int j0 = 0; j0 <= q0; j0 += 32) {
        // Prefetch next key tile while this one is consumed (global_prefetch_b8).
        if (j0 + 32 <= q0) {
            __builtin_prefetch(k16 + (bh * Ss + j0 + 32 + lane) * HDd, 0, 1);
            __builtin_prefetch(v16t + (bh * HDd + lane * 2) * Ss + j0 + 32, 0, 1);
        }

        // K B-fragments for this key tile.
        v16h bK[2][2];
#pragma unroll
        for (int ni = 0; ni < 2; ++ni)
#pragma unroll
            for (int kc = 0; kc < 2; ++kc)
                bK[ni][kc] = load_b_frag_vec(k16 + (bh * Ss + j0 + ni * 16) * HDd + kc * 32, HDd);

        // Scores: S = Q K^T (f32 accum).
        v8f sf[2][2];
#pragma unroll
        for (int mi = 0; mi < 2; ++mi)
#pragma unroll
            for (int ni = 0; ni < 2; ++ni) {
                v8f c = vzero8();
                c = WMMA_F16(aQ[mi][0], bK[ni][0], c);
                c = WMMA_F16(aQ[mi][1], bK[ni][1], c);
                sf[mi][ni] = c;
            }

        const bool diag = (j0 == q0);
#pragma unroll
        for (int mi = 0; mi < 2; ++mi)
#pragma unroll
            for (int ni = 0; ni < 2; ++ni)
#pragma unroll
                for (int r = 0; r < 8; ++r) {
                    float v = sf[mi][ni][r] * scale;
                    if (diag) {
                        const int qr = mi * 16 + r + g * 8;   // row within 32-row tile
                        const int kc = ni * 16 + nl;          // col within 32-key tile
                        if (kc > qr) v = -3.0e38f;            // causal mask
                    }
                    sf[mi][ni][r] = v;
                }

        // Online softmax per 16-row fragment group (row lives in 16 lanes of one half).
#pragma unroll
        for (int mi = 0; mi < 2; ++mi) {
            float rm[8], rs[8], corr[8];
#pragma unroll
            for (int r = 0; r < 8; ++r) rm[r] = fmaxf(sf[mi][0][r], sf[mi][1][r]);
#pragma unroll
            for (int msk = 1; msk <= 8; msk <<= 1)
#pragma unroll
                for (int r = 0; r < 8; ++r) rm[r] = fmaxf(rm[r], __shfl_xor(rm[r], msk, 32));
#pragma unroll
            for (int r = 0; r < 8; ++r) {
                const float mn = fmaxf(mrun[mi][r], rm[r]);
                corr[r] = __expf(mrun[mi][r] - mn);
                mrun[mi][r] = mn;
            }
#pragma unroll
            for (int ni = 0; ni < 2; ++ni)
#pragma unroll
                for (int r = 0; r < 8; ++r) sf[mi][ni][r] = __expf(sf[mi][ni][r] - mrun[mi][r]);
#pragma unroll
            for (int r = 0; r < 8; ++r) rs[r] = sf[mi][0][r] + sf[mi][1][r];
#pragma unroll
            for (int msk = 1; msk <= 8; msk <<= 1)
#pragma unroll
                for (int r = 0; r < 8; ++r) rs[r] += __shfl_xor(rs[r], msk, 32);
#pragma unroll
            for (int r = 0; r < 8; ++r) lrun[mi][r] = lrun[mi][r] * corr[r] + rs[r];
#pragma unroll
            for (int nv = 0; nv < 4; ++nv)
#pragma unroll
                for (int r = 0; r < 8; ++r) o[mi][nv][r] *= corr[r];
            // C-layout -> LDS (row-major P tile).
#pragma unroll
            for (int ni = 0; ni < 2; ++ni)
#pragma unroll
                for (int r = 0; r < 8; ++r)
                    Ps[(mi * 16 + r + g * 8) * 34 + ni * 16 + nl] = (_Float16)sf[mi][ni][r];
        }
        __syncthreads();  // single-wave workgroup: barrier ~ S_NOP; orders DS ops

        // P A-fragments + V B-fragments; O += P V.
        v16h aP[2];
        aP[0] = load_a_frag_f16(Ps, 34);
        aP[1] = load_a_frag_f16(Ps + 16 * 34, 34);
        v16h bV[4];
#pragma unroll
        for (int nv = 0; nv < 4; ++nv)
            bV[nv] = load_b_frag_vec(v16t + (bh * HDd + nv * 16) * Ss + j0, Ss);
#pragma unroll
        for (int mi = 0; mi < 2; ++mi)
#pragma unroll
            for (int nv = 0; nv < 4; ++nv) o[mi][nv] = WMMA_F16(aP[mi], bV[nv], o[mi][nv]);
        __syncthreads();
    }

    // Epilogue: normalize and write fp32 output [b, s, h*64+hd].
#pragma unroll
    for (int mi = 0; mi < 2; ++mi) {
        float inv[8];
#pragma unroll
        for (int r = 0; r < 8; ++r) inv[r] = 1.0f / lrun[mi][r];
#pragma unroll
        for (int nv = 0; nv < 4; ++nv)
#pragma unroll
            for (int r = 0; r < 8; ++r) {
                const int qr = q0 + mi * 16 + r + g * 8;
                out[((long)b * Ss + qr) * Dd + h * HDd + nv * 16 + nl] = o[mi][nv][r] * inv[r];
            }
    }
}

extern "C" void kernel_launch(void* const* d_in, const int* in_sizes, int n_in,
                              void* d_out, int out_size, void* d_ws, size_t ws_size,
                              hipStream_t stream) {
    const float* qx = (const float*)d_in[0];
    const float* kx = (const float*)d_in[1];
    const float* vx = (const float*)d_in[2];
    // d_in[3] = causal mask (int), applied analytically instead.
    const float* Wq = (const float*)d_in[4];
    const float* bq = (const float*)d_in[5];
    const float* Wk = (const float*)d_in[6];
    const float* bk = (const float*)d_in[7];
    const float* Wv = (const float*)d_in[8];
    const float* bv = (const float*)d_in[9];
    float* out = (float*)d_out;

    const long N = (long)Bb * Hh * Ss * HDd;  // 4M elements per tensor
    _Float16* q16  = (_Float16*)d_ws;
    _Float16* k16  = q16 + N;
    _Float16* v16t = q16 + 2 * N;             // needs 24 MB of workspace

    dim3 gp((Bb * Ss) / 128, Dd / 64, 3);     // (32, 16, 3)
    qkv_proj_kernel<<<gp, dim3(256), PROJ_LDS, stream>>>(qx, kx, vx, Wq, bq, Wk, bk, Wv, bv,
                                                         q16, k16, v16t);

    dim3 ga(Ss / 32, Hh, Bb);                 // (64, 16, 2)
    attn_kernel<<<ga, dim3(32), 0, stream>>>(q16, k16, v16t, out);
}
